// MultiHeadGraphAttention_89395449299815
// MI455X (gfx1250) — matrix operand order repaired and checked by reading, not compiled, not run on hardware
//
#include <hip/hip_runtime.h>
#include <hip/hip_bf16.h>
#include <stdint.h>

// Shapes fixed by setup_inputs(): B=2, N=2048, D=256, H=8, dh=32
#define BN   2
#define NN   2048
#define DD   256
#define HH   8
#define DH   32

typedef float  v2f   __attribute__((ext_vector_type(2)));
typedef float  v8f   __attribute__((ext_vector_type(8)));
typedef __bf16 v16bf __attribute__((ext_vector_type(16)));
typedef int    v2i   __attribute__((ext_vector_type(2)));
typedef __attribute__((address_space(1))) v2i* as1_v2i;
typedef __attribute__((address_space(3))) v2i* as3_v2i;

#if defined(__has_builtin)
#if __has_builtin(__builtin_amdgcn_global_load_async_to_lds_b64)
#define HAS_ASYNC_LDS 1
#endif
#endif

// ---------------------------------------------------------------------------
// Kernel 1: x_proj = x @ Wf  (f32 WMMA 16x16x4), stored transposed per head:
//   Vt [b, h, f, n]  f32   (for e_src/e_dst precision)
//   Vtb[b, h, f, n]  bf16  (for the attention aggregation WMMA)
// M = B*N = 4096 rows, N' = H*dh = 256 cols, K = D = 256.
// One 16x16 tile per wave; 8 waves/block; 512 blocks.
// ---------------------------------------------------------------------------
__global__ __launch_bounds__(256) void proj_kernel(
    const float* __restrict__ x, const float* __restrict__ W,
    float* __restrict__ Vt, __bf16* __restrict__ Vtb)
{
    const int lane = threadIdx.x & 31;
    const int tile = blockIdx.x * 8 + (threadIdx.x >> 5);   // 0..4095
    const int tm = tile >> 4;      // 0..255  (16-row tile of M)
    const int tn = tile & 15;      // 0..15   (16-col tile of N')
    const int row  = lane & 15;
    const int half = lane >> 4;

    // A operand (16x4 f32): lane<16 -> K pair (k,k+1); lane>=16 -> (k+2,k+3)
    const float* arow = x + (size_t)(tm * 16 + row) * DD;
    // B operand (4x16 f32): col c = tn*16 + (lane&15); Wf[k,c] = W[c>>5][k][c&31]
    const int c = tn * 16 + row;
    const float* bcol = W + (size_t)(c >> 5) * (DD * DH) + (c & 31);

    v8f acc = {};
    #pragma unroll 8
    for (int k = 0; k < DD; k += 4) {
        const int kb = k + half * 2;
        v2f a = *(const v2f*)(arow + kb);          // 8B aligned (kb even)
        v2f b; b.x = bcol[(size_t)kb * DH]; b.y = bcol[(size_t)(kb + 1) * DH];
        acc = __builtin_amdgcn_wmma_f32_16x16x4_f32(
                  false, a, false, b, (short)0, acc, false, false);
    }

    // C layout: VGPR r -> row m = r + half*8, col = lane&15. Transposed store:
    // consecutive r -> consecutive n, so the 8 stores coalesce into b128s.
    #pragma unroll
    for (int r = 0; r < 8; ++r) {
        const int m = tm * 16 + r + half * 8;
        const int b = m >> 11, n = m & (NN - 1);
        const int h = c >> 5, f = c & 31;
        const size_t idx = ((size_t)((b * HH + h) * DH + f) << 11) + n;
        Vt[idx]  = acc[r];
        Vtb[idx] = (__bf16)acc[r];
    }
}

// ---------------------------------------------------------------------------
// Kernel 2: e_src[b,h,n] = sum_f Vt[b,h,f,n]*a_src[h,f];  same for e_dst.
// ---------------------------------------------------------------------------
__global__ __launch_bounds__(256) void edge_kernel(
    const float* __restrict__ Vt, const float* __restrict__ a_src,
    const float* __restrict__ a_dst, float* __restrict__ es,
    float* __restrict__ ed)
{
    const int idx = blockIdx.x * blockDim.x + threadIdx.x;  // B*H*N = 32768
    const int n = idx & (NN - 1);
    const int h = (idx >> 11) & (HH - 1);
    const int b = idx >> 14;
    const float* base = Vt + ((size_t)((b * HH + h) * DH) << 11) + n;
    float s = 0.f, d = 0.f;
    #pragma unroll
    for (int f = 0; f < DH; ++f) {
        const float v = base[(size_t)f << 11];
        s += v * a_src[h * DH + f];
        d += v * a_dst[h * DH + f];
    }
    es[idx] = s;
    ed[idx] = d;
}

// ---------------------------------------------------------------------------
// Kernel 3: fused flash-attention over j, per (b,h, 128-row i-block):
//   score  = leaky_relu(e_src[i]+e_dst[j]) + beta*log(prior+eps), adj-masked
//   online softmax; acc += P(bf16) x V(bf16) via v_wmma_f32_16x16x32_bf16
// The 32x32 bf16 V tile is staged once per block in LDS (async copy where
// available) and shared by all 8 waves -> 8x cut in L2 V-traffic.
// LDS layout: tile[f][j], row stride 40 bf16 (80B) to spread banks.
// Lane layout of scores == WMMA A-operand layout (16x32 bf16):
//   lane<16 : row = lane,    K = 0..15
//   lane>=16: row = lane-16, K = 16..31
// Grid: B*H*(N/128) = 256 blocks, 8 waves each (one 16-row i-tile per wave).
// ---------------------------------------------------------------------------
#define TSTRIDE 40

__global__ __launch_bounds__(256) void attn_kernel(
    const float* __restrict__ prior, const int* __restrict__ adj,
    const __bf16* __restrict__ Vtb, const float* __restrict__ es,
    const float* __restrict__ ed, const float* __restrict__ beta_tilde,
    float* __restrict__ hagg)
{
    __shared__ __align__(16) __bf16 tile[32 * TSTRIDE];

    const int lane = threadIdx.x & 31;
    const int wave = threadIdx.x >> 5;
    const int bid = blockIdx.x;
    const int ib = bid & 15;
    const int h  = (bid >> 4) & (HH - 1);
    const int b  = bid >> 7;
    const int i0 = ib * 128 + wave * 16;

    const int row  = lane & 15;
    const int half = lane >> 4;
    const int jb   = half * 8;
    const int i    = i0 + row;

    // cooperative V-tile fill coords: 256 threads x 4 bf16 (8B) = 2KB tile
    const int tf = threadIdx.x >> 3;        // f row 0..31
    const int tc = (threadIdx.x & 7) * 4;   // j chunk 0,4,..,28

    const float beta = log1pf(__expf(beta_tilde[0]));
    const float esrc = es[((b * HH + h) << 11) + i];

    const float* prow  = prior + ((size_t)b << 22) + ((size_t)i << 11);
    const int*   arow  = adj + ((size_t)i << 11);
    const float* edrow = ed + ((size_t)(b * HH + h) << 11);
    const __bf16* vtb  = Vtb + ((size_t)((b * HH + h) * DH) << 11);

    float m = -1e30f, lsum = 0.f;
    v8f acc0 = {}, acc1 = {};

    for (int jt = 0; jt < NN; jt += 32) {
        __syncthreads();   // all waves done reading previous tile
        // ---- stage V tile (global bf16 -> LDS), overlapped with score math
        {
            const __bf16* gsrc = vtb + ((size_t)tf << 11) + jt + tc;
            __bf16* ldst = &tile[tf * TSTRIDE + tc];
#ifdef HAS_ASYNC_LDS
            // param0: global addrspace(1) v2i*, param1: LDS addrspace(3) v2i*
            // (integer casts: as1 shares the 64-bit flat value; as3 pointer is
            //  the low 32 bits of the flat address = LDS offset per aperture)
            __builtin_amdgcn_global_load_async_to_lds_b64(
                (as1_v2i)(unsigned long long)gsrc,
                (as3_v2i)(unsigned int)(unsigned long long)ldst, 0, 0);
#else
            *(uint2*)ldst = *(const uint2*)gsrc;
#endif
        }

        // prefetch next prior/adj tile into cache (global_prefetch_b8)
        __builtin_prefetch(prow + jt + 32 + jb, 0, 0);
        __builtin_prefetch(arow + jt + 32 + jb, 0, 0);

        float s[16];
        float tmax = -1e30f;
        #pragma unroll
        for (int e = 0; e < 16; ++e) {
            const int j = jt + half * 16 + e;
            float t = esrc + edrow[j];
            t = (t > 0.f) ? t : 0.2f * t;                 // leaky_relu
            t += beta * __logf(prow[j] + 1e-6f);
            s[e] = (arow[j] != 0) ? t : -1e30f;
            tmax = fmaxf(tmax, s[e]);
        }
        // full-row max: lanes l and l^16 together cover j 0..31 of the row
        tmax = fmaxf(tmax, __shfl_xor(tmax, 16, 32));
        const float mnew  = fmaxf(m, tmax);
        const float scale = __expf(m - mnew);

        float tsum = 0.f;
        v16bf pa;
        #pragma unroll
        for (int e = 0; e < 16; ++e) {
            const float p = __expf(s[e] - mnew);
            tsum += p;
            pa[e] = (__bf16)p;
        }
        tsum += __shfl_xor(tsum, 16, 32);
        lsum = lsum * scale + tsum;
        m = mnew;

        // rescale accumulators: C VGPR r holds row (r + half*8)
        #pragma unroll
        for (int r = 0; r < 8; ++r) {
            const float sr = __shfl(scale, r + half * 8, 32);
            acc0[r] *= sr;
            acc1[r] *= sr;
        }

#ifdef HAS_ASYNC_LDS
#if __has_builtin(__builtin_amdgcn_s_wait_asynccnt)
        __builtin_amdgcn_s_wait_asynccnt(0);
#else
        asm volatile("s_wait_asynccnt 0x0" ::: "memory");
#endif
#endif
        __syncthreads();   // tile fully staged & visible

        // B operands (32x16 bf16) from LDS: elem e -> K = half*16 + e,
        // col f = g*16 + (lane&15); 32B contiguous per lane.
        v16bf vb0 = *(const v16bf*)&tile[row * TSTRIDE + half * 16];
        v16bf vb1 = *(const v16bf*)&tile[(16 + row) * TSTRIDE + half * 16];

        acc0 = __builtin_amdgcn_wmma_f32_16x16x32_bf16(
                   false, pa, false, vb0, (short)0, acc0, false, false);
        acc1 = __builtin_amdgcn_wmma_f32_16x16x32_bf16(
                   false, pa, false, vb1, (short)0, acc1, false, false);
    }

    const float linv = 1.0f / lsum;
    #pragma unroll
    for (int r = 0; r < 8; ++r) {
        const float lr = __shfl(linv, r + half * 8, 32);
        const int ii = i0 + r + half * 8;
        float* orow = hagg + ((size_t)(b * NN + ii) << 8) + h * DH;
        orow[row]      = acc0[r] * lr;
        orow[16 + row] = acc1[r] * lr;
    }
}

// ---------------------------------------------------------------------------
// Kernel 4: out = hagg @ W_out^T   (f32 WMMA 16x16x4)
// out[m,d] = sum_c hagg[m,c] * W_out[d,c]; B-operand pairs contiguous in c.
// ---------------------------------------------------------------------------
__global__ __launch_bounds__(256) void outgemm_kernel(
    const float* __restrict__ hagg, const float* __restrict__ W_out,
    float* __restrict__ out)
{
    const int lane = threadIdx.x & 31;
    const int tile = blockIdx.x * 8 + (threadIdx.x >> 5);
    const int tm = tile >> 4;
    const int tn = tile & 15;
    const int row  = lane & 15;
    const int half = lane >> 4;

    const float* arow = hagg + (size_t)(tm * 16 + row) * DD;
    const int d = tn * 16 + row;
    const float* bcol = W_out + (size_t)d * DD;   // contiguous in k (=c)

    v8f acc = {};
    #pragma unroll 8
    for (int k = 0; k < DD; k += 4) {
        const int kb = k + half * 2;
        v2f a = *(const v2f*)(arow + kb);
        v2f b = *(const v2f*)(bcol + kb);
        acc = __builtin_amdgcn_wmma_f32_16x16x4_f32(
                  false, a, false, b, (short)0, acc, false, false);
    }
    #pragma unroll
    for (int r = 0; r < 8; ++r) {
        const int mrow = tm * 16 + r + half * 8;
        out[(size_t)mrow * DD + tn * 16 + row] = acc[r];
    }
}

// ---------------------------------------------------------------------------
extern "C" void kernel_launch(void* const* d_in, const int* in_sizes, int n_in,
                              void* d_out, int out_size, void* d_ws, size_t ws_size,
                              hipStream_t stream) {
    const float* x          = (const float*)d_in[0];
    const int*   adj        = (const int*)  d_in[1];
    const float* prior      = (const float*)d_in[2];
    const float* W          = (const float*)d_in[3];
    const float* a_src      = (const float*)d_in[4];
    const float* a_dst      = (const float*)d_in[5];
    const float* beta_tilde = (const float*)d_in[6];
    const float* W_out      = (const float*)d_in[7];
    float* out = (float*)d_out;

    // workspace layout
    float*  ws   = (float*)d_ws;
    float*  Vt   = ws;                                  // 1,048,576 f32
    float*  es   = Vt + (size_t)BN * HH * DH * NN;      // 32,768 f32
    float*  ed   = es + (size_t)BN * HH * NN;           // 32,768 f32
    float*  hagg = ed + (size_t)BN * HH * NN;           // 1,048,576 f32
    __bf16* Vtb  = (__bf16*)(hagg + (size_t)BN * NN * DD); // 1,048,576 bf16

    proj_kernel<<<512, 256, 0, stream>>>(x, W, Vt, Vtb);
    edge_kernel<<<(BN * HH * NN) / 256, 256, 0, stream>>>(Vt, a_src, a_dst, es, ed);
    attn_kernel<<<BN * HH * (NN / 128), 256, 0, stream>>>(prior, adj, Vtb, es, ed,
                                                          beta_tilde, hagg);
    outgemm_kernel<<<512, 256, 0, stream>>>(hagg, W_out, out);
}
